// ScaledDPattn_73778948211335
// MI455X (gfx1250) — matrix operand order repaired and checked by reading, not compiled, not run on hardware
//
#include <hip/hip_runtime.h>

typedef __attribute__((ext_vector_type(16))) _Float16 v16h;
typedef __attribute__((ext_vector_type(8)))  float    v8f;
typedef __attribute__((ext_vector_type(4)))  unsigned v4u;
typedef __attribute__((ext_vector_type(4)))  int      v4i;
typedef __attribute__((ext_vector_type(8)))  int      v8i;

#define S_LEN 2048
#define D_DIM 64
#define BH    32            // B*H
#define WAVES 4
#define BLOCK (WAVES * 32)
#define QT    16            // query rows per wave
#define KC    64            // keys per chunk
#define NCH   (S_LEN / KC)  // 32 chunks
#define KSTR  72            // padded LDS row stride (halves): 32dw + 4dw pad, 16B-aligned rows
#define VSTR  72

union HFrag { v16h v; _Float16 h[16]; };
union FAcc  { v8f  v; float    f[8];  };

// ---- 16-half fragment loads as two 16B (b128) LDS accesses -------------------
__device__ __forceinline__ v16h load_frag16(const _Float16* p) {
    union { v16h v; uint4 q[2]; } u;
    u.q[0] = *(const uint4*)(p);
    u.q[1] = *(const uint4*)(p + 8);
    return u.v;
}
__device__ __forceinline__ v16h load_frag16_2(const _Float16* p0, const _Float16* p1) {
    union { v16h v; uint4 q[2]; } u;
    u.q[0] = *(const uint4*)(p0);
    u.q[1] = *(const uint4*)(p1);
    return u.v;
}

// ---------------- width-16 xor reductions via v_permlane16_b32 ----------------
__device__ __forceinline__ float permx16(float x, int sel0, int sel1) {
    int xi = __builtin_bit_cast(int, x);
    int r  = __builtin_amdgcn_permlane16(xi, xi, sel0, sel1, false, false);
    return __builtin_bit_cast(float, r);
}
__device__ __forceinline__ float rowmax16(float x) {
    x = fmaxf(x, permx16(x, 0x67452301, 0xEFCDAB89));  // xor 1
    x = fmaxf(x, permx16(x, 0x54761032, 0xDCFE98BA));  // xor 2
    x = fmaxf(x, permx16(x, 0x32107654, 0xBA98FEDC));  // xor 4
    x = fmaxf(x, permx16(x, 0xFEDCBA98, 0x76543210));  // xor 8
    return x;
}

// ---------------- TDM: DMA one contiguous 8KB f16 tile into LDS ---------------
// D# per CDNA5 ISA ch.8: 1-row tile of 1024 x 8B units, pad_enable inserts
// 4 dwords of LDS padding every 32 dwords (=> 72-half padded row stride,
// keeping every row 16B-aligned for b128 fragment loads).
__device__ __forceinline__ void tdm_load_tile(unsigned lds_off, const void* gptr) {
    unsigned long long ga = (unsigned long long)gptr;
    v4u g0;
    g0[0] = 1u;                                            // count=1, user desc
    g0[1] = lds_off;                                       // lds_addr (bytes)
    g0[2] = (unsigned)(ga & 0xFFFFFFFFu);                  // global_addr[31:0]
    g0[3] = (unsigned)((ga >> 32) & 0x01FFFFFFu)           // global_addr[56:32]
          | (2u << 30);                                    // type=2 ("image")
    v8i g1;
    g1[0] = (3 << 16)        // data_size = 8 bytes
          | (1 << 20)        // pad_enable
          | (4 << 22)        // pad_interval code 4 = every 32 dwords (128B row)
          | (3 << 25);       // pad_amount  code 3 = 4 dwords (16B)
    g1[1] = (1024 & 0xFFFF) << 16;          // tensor_dim0[15:0]  (bits 63:48)
    g1[2] = (0) | (1 << 16);                // tensor_dim0 hi, tensor_dim1=1
    g1[3] = (0) | ((1024 & 0xFFFF) << 16);  // tensor_dim1 hi, tile_dim0=1024
    g1[4] = 1;                              // tile_dim1=1, tile_dim2=0
    g1[5] = 1024;                           // tensor_dim0_stride lo
    g1[6] = 0;
    g1[7] = 0;
    v4i z4 = {0, 0, 0, 0};
#if __clang_major__ >= 23
    v8i z8 = {0, 0, 0, 0, 0, 0, 0, 0};
    __builtin_amdgcn_tensor_load_to_lds(g0, g1, z4, z4, z8, 0);
#else
    __builtin_amdgcn_tensor_load_to_lds(g0, g1, z4, z4, 0);
#endif
}

__device__ __forceinline__ unsigned lds_offset_of(const void* p) {
    return (unsigned)(size_t)p;   // low 32 bits of a flat LDS pointer = LDS byte addr
}

// =============== pre-pass: K -> f16 (same layout), V -> f16 transposed tiles ==
// Vt layout: [bh][chunk][d:64][key:64] f16, each chunk tile contiguous (8KB)
__global__ __launch_bounds__(256)
void ScaledDPattn_convert_kv(const float* __restrict__ k,
                             const float* __restrict__ v,
                             _Float16* __restrict__ kh,
                             _Float16* __restrict__ vt)
{
    size_t i4 = ((size_t)blockIdx.x * 256 + threadIdx.x) * 4;
    float4 kf = *(const float4*)(k + i4);
    kh[i4 + 0] = (_Float16)kf.x;
    kh[i4 + 1] = (_Float16)kf.y;
    kh[i4 + 2] = (_Float16)kf.z;
    kh[i4 + 3] = (_Float16)kf.w;

    float4 vf = *(const float4*)(v + i4);
    float vv[4] = { vf.x, vf.y, vf.z, vf.w };
    size_t key_g = i4 >> 6;                 // global row index (bh*S + key)
    int bh    = (int)(key_g >> 11);
    int key   = (int)(key_g & (S_LEN - 1));
    int chunk = key >> 6;
    int kk    = key & (KC - 1);
    int d0    = (int)(i4 & (D_DIM - 1));
    #pragma unroll
    for (int c = 0; c < 4; ++c)
        vt[((((size_t)bh * NCH + chunk) * D_DIM) + d0 + c) * KC + kk] = (_Float16)vv[c];
}

// ====================== main kernel: TDM-staged flash attention ===============
__global__ __launch_bounds__(BLOCK)
void ScaledDPattn_fa_wmma_tdm(const float* __restrict__ q,
                              const _Float16* __restrict__ kh,
                              const _Float16* __restrict__ vt,
                              float* __restrict__ out)
{
    __shared__ __align__(16) _Float16 Kb[2][KC * KSTR];    // [key][d], padded
    __shared__ __align__(16) _Float16 Vb[2][D_DIM * VSTR]; // [d][key], padded
    __shared__ __align__(16) _Float16 Plds[WAVES * QT * KC];

    const int tid  = threadIdx.x;
    const int lane = tid & 31;
    const int wave = tid >> 5;
    const int bh   = blockIdx.y;
    const int q0   = blockIdx.x * (WAVES * QT) + wave * QT;

    const int half = lane >> 4;
    const int l16  = lane & 15;

    const _Float16* kh_base = kh + (size_t)bh * S_LEN * D_DIM;
    const _Float16* vt_base = vt + (size_t)bh * NCH * (KC * D_DIM);

    // pre-issue TDM loads for chunks 0 and 1 (double buffered)
    if (wave == 0) {
        tdm_load_tile(lds_offset_of(&Kb[0][0]), kh_base + 0 * KC * D_DIM);
        tdm_load_tile(lds_offset_of(&Vb[0][0]), vt_base + 0 * KC * D_DIM);
        tdm_load_tile(lds_offset_of(&Kb[1][0]), kh_base + 1 * KC * D_DIM);
        tdm_load_tile(lds_offset_of(&Vb[1][0]), vt_base + 1 * KC * D_DIM);
    }

    // Q scale folds 1/sqrt(64) AND log2(e): softmax runs in the exp2 domain.
    // A-fragment element i holds K-index kb+i (i<8) / 16+kb+i-8 (i>=8), kb=8*half.
    const float scale = 0.125f * 1.44269504088896340736f;
    const float* qrow = q + ((size_t)bh * S_LEN + q0 + l16) * D_DIM;
    HFrag aQ[2];
    #pragma unroll
    for (int f = 0; f < 2; ++f)
        #pragma unroll
        for (int i = 0; i < 16; ++i) {
            int kk = (i < 8) ? (half * 8 + i) : (16 + half * 8 + (i - 8));
            aQ[f].h[i] = (_Float16)(qrow[f * 32 + kk] * scale);
        }

    HFrag bOnes;
    #pragma unroll
    for (int i = 0; i < 16; ++i) bOnes.h[i] = (_Float16)1.0f;

    FAcc acc[5];                 // acc[0..3] = O tiles, acc[4] = row-sum l
    #pragma unroll
    for (int n = 0; n < 5; ++n)
        #pragma unroll
        for (int i = 0; i < 8; ++i) acc[n].f[i] = 0.0f;
    float mrow[8];
    #pragma unroll
    for (int i = 0; i < 8; ++i) mrow[i] = -1e30f;

    _Float16* Pw = Plds + wave * (QT * KC);

    for (int chunk = 0; chunk < NCH; ++chunk) {
        const int buf = chunk & 1;
        if (wave == 0) {
            if (chunk + 1 < NCH) __builtin_amdgcn_s_wait_tensorcnt(2);
            else                 __builtin_amdgcn_s_wait_tensorcnt(0);
        }
        __syncthreads();                       // chunk's K/V tiles visible

        // ---- S = Q * K^T (log2 domain), four 16-key subtiles
        FAcc sc[4];
        #pragma unroll
        for (int t = 0; t < 4; ++t) {
            const _Float16* kr = &Kb[buf][(t * 16 + l16) * KSTR + half * 8];
            v16h bK0 = load_frag16_2(kr, kr + 16);        // d = 0..31  (per lane half)
            v16h bK1 = load_frag16_2(kr + 32, kr + 48);   // d = 32..63
            v8f c = {};
            c = __builtin_amdgcn_wmma_f32_16x16x32_f16(false, aQ[0].v, false, bK0,
                                                       (short)0, c, false, false);
            c = __builtin_amdgcn_wmma_f32_16x16x32_f16(false, aQ[1].v, false, bK1,
                                                       (short)0, c, false, false);
            sc[t].v = c;
        }

        // ---- online softmax: row max via permlane16, exp2, rescale
        float alpha[8];
        #pragma unroll
        for (int i = 0; i < 8; ++i) {
            float x = fmaxf(fmaxf(sc[0].f[i], sc[1].f[i]),
                            fmaxf(sc[2].f[i], sc[3].f[i]));
            x = rowmax16(x);
            float mn = fmaxf(mrow[i], x);
            alpha[i] = __builtin_amdgcn_exp2f(mrow[i] - mn);
            mrow[i]  = mn;
            #pragma unroll
            for (int t = 0; t < 4; ++t)
                sc[t].f[i] = __builtin_amdgcn_exp2f(sc[t].f[i] - mn);
        }
        #pragma unroll
        for (int n = 0; n < 5; ++n)
            #pragma unroll
            for (int i = 0; i < 8; ++i) acc[n].f[i] *= alpha[i];

        // ---- C-layout P -> A-layout via per-wave LDS scratch
        #pragma unroll
        for (int t = 0; t < 4; ++t)
            #pragma unroll
            for (int i = 0; i < 8; ++i)
                Pw[(i + half * 8) * KC + t * 16 + l16] = (_Float16)sc[t].f[i];
        asm volatile("" ::: "memory");

        HFrag aP[2];
        #pragma unroll
        for (int f = 0; f < 2; ++f) {
            const _Float16* p0 = Pw + l16 * KC + f * 32 + half * 8;
            aP[f].v = load_frag16_2(p0, p0 + 16);
        }

        // ---- O += P * V + row-sum via ones-WMMA
        #pragma unroll
        for (int n = 0; n < 4; ++n)
            #pragma unroll
            for (int f = 0; f < 2; ++f) {
                v16h bV = load_frag16_2(&Vb[buf][(n * 16 + l16) * VSTR + f * 32 + half * 8], &Vb[buf][(n * 16 + l16) * VSTR + f * 32 + half * 8 + 16]);
                acc[n].v = __builtin_amdgcn_wmma_f32_16x16x32_f16(false, aP[f].v, false, bV,
                                                                  (short)0, acc[n].v,
                                                                  false, false);
            }
        acc[4].v = __builtin_amdgcn_wmma_f32_16x16x32_f16(false, aP[0].v, false, bOnes.v,
                                                          (short)0, acc[4].v, false, false);
        acc[4].v = __builtin_amdgcn_wmma_f32_16x16x32_f16(false, aP[1].v, false, bOnes.v,
                                                          (short)0, acc[4].v, false, false);
        __syncthreads();                       // all waves done with buf

        if (wave == 0 && chunk + 2 < NCH) {    // refill freed buffer, 2 ahead
            tdm_load_tile(lds_offset_of(&Kb[buf][0]), kh_base + (chunk + 2) * KC * D_DIM);
            tdm_load_tile(lds_offset_of(&Vb[buf][0]), vt_base + (chunk + 2) * KC * D_DIM);
        }
    }

    float* orow = out + ((size_t)bh * S_LEN + q0) * D_DIM;
    #pragma unroll
    for (int i = 0; i < 8; ++i) {
        float rl = __builtin_amdgcn_rcpf(acc[4].f[i]);
        int m = i + half * 8;
        #pragma unroll
        for (int n = 0; n < 4; ++n)
            orow[(size_t)m * D_DIM + n * 16 + l16] = acc[n].f[i] * rl;
    }
}

// ====================== fallback (no workspace), padded staging ===============
__global__ __launch_bounds__(BLOCK)
void ScaledDPattn_fa_wmma_fb(const float* __restrict__ q,
                             const float* __restrict__ k,
                             const float* __restrict__ v,
                             float* __restrict__ out)
{
    __shared__ __align__(16) _Float16 Klds[KC * KSTR];
    __shared__ __align__(16) _Float16 Vlds[D_DIM * VSTR];
    __shared__ __align__(16) _Float16 Plds[WAVES * QT * KC];

    const int tid  = threadIdx.x;
    const int lane = tid & 31;
    const int wave = tid >> 5;
    const int bh   = blockIdx.y;
    const int q0   = blockIdx.x * (WAVES * QT) + wave * QT;
    const int half = lane >> 4;
    const int l16  = lane & 15;

    const float scale = 0.125f * 1.44269504088896340736f;
    const float* qrow = q + ((size_t)bh * S_LEN + q0 + l16) * D_DIM;
    HFrag aQ[2];
    #pragma unroll
    for (int f = 0; f < 2; ++f)
        #pragma unroll
        for (int i = 0; i < 16; ++i) {
            int kk = (i < 8) ? (half * 8 + i) : (16 + half * 8 + (i - 8));
            aQ[f].h[i] = (_Float16)(qrow[f * 32 + kk] * scale);
        }
    HFrag bOnes;
    #pragma unroll
    for (int i = 0; i < 16; ++i) bOnes.h[i] = (_Float16)1.0f;

    FAcc acc[5];
    #pragma unroll
    for (int n = 0; n < 5; ++n)
        #pragma unroll
        for (int i = 0; i < 8; ++i) acc[n].f[i] = 0.0f;
    float mrow[8];
    #pragma unroll
    for (int i = 0; i < 8; ++i) mrow[i] = -1e30f;

    const float* kg_base = k + (size_t)bh * S_LEN * D_DIM;
    const float* vg_base = v + (size_t)bh * S_LEN * D_DIM;
    _Float16* Pw = Plds + wave * (QT * KC);

    for (int chunk = 0; chunk < NCH; ++chunk) {
        const float* kg = kg_base + chunk * KC * D_DIM;
        const float* vg = vg_base + chunk * KC * D_DIM;
        #pragma unroll
        for (int it = 0; it < (KC * D_DIM) / (BLOCK * 4); ++it) {
            int e = tid * 4 + it * BLOCK * 4;
            int row = e >> 6, d = e & 63;
            float4 kf = *(const float4*)(kg + e);
            float4 vf = *(const float4*)(vg + e);
            Klds[row * KSTR + d + 0] = (_Float16)kf.x;
            Klds[row * KSTR + d + 1] = (_Float16)kf.y;
            Klds[row * KSTR + d + 2] = (_Float16)kf.z;
            Klds[row * KSTR + d + 3] = (_Float16)kf.w;
            float vv[4] = { vf.x, vf.y, vf.z, vf.w };
            #pragma unroll
            for (int c = 0; c < 4; ++c)
                Vlds[(d + c) * VSTR + row] = (_Float16)vv[c];
            __builtin_prefetch(kg + KC * D_DIM + e, 0, 1);
            __builtin_prefetch(vg + KC * D_DIM + e, 0, 1);
        }
        __syncthreads();

        FAcc sc[4];
        #pragma unroll
        for (int t = 0; t < 4; ++t) {
            const _Float16* kr = Klds + (t * 16 + l16) * KSTR + half * 8;
            v16h bK0 = load_frag16_2(kr, kr + 16);
            v16h bK1 = load_frag16_2(kr + 32, kr + 48);
            v8f c = {};
            c = __builtin_amdgcn_wmma_f32_16x16x32_f16(false, aQ[0].v, false, bK0,
                                                       (short)0, c, false, false);
            c = __builtin_amdgcn_wmma_f32_16x16x32_f16(false, aQ[1].v, false, bK1,
                                                       (short)0, c, false, false);
            sc[t].v = c;
        }
        float alpha[8];
        #pragma unroll
        for (int i = 0; i < 8; ++i) {
            float x = fmaxf(fmaxf(sc[0].f[i], sc[1].f[i]),
                            fmaxf(sc[2].f[i], sc[3].f[i]));
            x = rowmax16(x);
            float mn = fmaxf(mrow[i], x);
            alpha[i] = __builtin_amdgcn_exp2f(mrow[i] - mn);
            mrow[i]  = mn;
            #pragma unroll
            for (int t = 0; t < 4; ++t)
                sc[t].f[i] = __builtin_amdgcn_exp2f(sc[t].f[i] - mn);
        }
        #pragma unroll
        for (int n = 0; n < 5; ++n)
            #pragma unroll
            for (int i = 0; i < 8; ++i) acc[n].f[i] *= alpha[i];

        #pragma unroll
        for (int t = 0; t < 4; ++t)
            #pragma unroll
            for (int i = 0; i < 8; ++i)
                Pw[(i + half * 8) * KC + t * 16 + l16] = (_Float16)sc[t].f[i];
        asm volatile("" ::: "memory");

        HFrag aP[2];
        #pragma unroll
        for (int f = 0; f < 2; ++f) {
            const _Float16* p0 = Pw + l16 * KC + f * 32 + half * 8;
            aP[f].v = load_frag16_2(p0, p0 + 16);
        }
        #pragma unroll
        for (int n = 0; n < 4; ++n)
            #pragma unroll
            for (int f = 0; f < 2; ++f) {
                v16h bV = load_frag16_2(Vlds + (n * 16 + l16) * VSTR + f * 32 + half * 8, Vlds + (n * 16 + l16) * VSTR + f * 32 + half * 8 + 16);
                acc[n].v = __builtin_amdgcn_wmma_f32_16x16x32_f16(false, aP[f].v, false, bV,
                                                                  (short)0, acc[n].v,
                                                                  false, false);
            }
        acc[4].v = __builtin_amdgcn_wmma_f32_16x16x32_f16(false, aP[0].v, false, bOnes.v,
                                                          (short)0, acc[4].v, false, false);
        acc[4].v = __builtin_amdgcn_wmma_f32_16x16x32_f16(false, aP[1].v, false, bOnes.v,
                                                          (short)0, acc[4].v, false, false);
        __syncthreads();
    }

    float* orow = out + ((size_t)bh * S_LEN + q0) * D_DIM;
    #pragma unroll
    for (int i = 0; i < 8; ++i) {
        float rl = __builtin_amdgcn_rcpf(acc[4].f[i]);
        int m = i + half * 8;
        #pragma unroll
        for (int n = 0; n < 4; ++n)
            orow[(size_t)m * D_DIM + n * 16 + l16] = acc[n].f[i] * rl;
    }
}

extern "C" void kernel_launch(void* const* d_in, const int* in_sizes, int n_in,
                              void* d_out, int out_size, void* d_ws, size_t ws_size,
                              hipStream_t stream) {
    const float* q = (const float*)d_in[0];
    const float* k = (const float*)d_in[1];
    const float* v = (const float*)d_in[2];
    float* out = (float*)d_out;

    const size_t elems   = (size_t)BH * S_LEN * D_DIM;      // 4,194,304
    const size_t ws_need = 2 * elems * sizeof(_Float16);    // K(f16) + Vt(f16)

    dim3 grid(S_LEN / (WAVES * QT), BH);
    if (ws_size >= ws_need) {
        _Float16* kh = (_Float16*)d_ws;
        _Float16* vt = kh + elems;
        ScaledDPattn_convert_kv<<<dim3((unsigned)(elems / (256 * 4))), dim3(256), 0, stream>>>(k, v, kh, vt);
        ScaledDPattn_fa_wmma_tdm<<<grid, dim3(BLOCK), 0, stream>>>(q, kh, vt, out);
    } else {
        ScaledDPattn_fa_wmma_fb<<<grid, dim3(BLOCK), 0, stream>>>(q, k, v, out);
    }
}
